// EGNNModel_16587163698062
// MI455X (gfx1250) — matrix operand order repaired and checked by reading, not compiled, not run on hardware
//
#include <hip/hip_runtime.h>
#include <hip/hip_bf16.h>
#include <math.h>

// ---------------------------------------------------------------------------
// EGNN forward for MI455X (gfx1250, wave32, WMMA 16x16x32 f16->f32).
// Edge MLP (dominant, ~1.26 TFLOP) runs on v_wmma_f32_16x16x32_f16 with
// fragment-swizzled LDS staging; segment sums via f32 global atomics
// (h/agg are L2-resident at 51MB each, 192MB L2).
// ---------------------------------------------------------------------------

typedef _Float16 v16h __attribute__((ext_vector_type(16)));
typedef float    v8f  __attribute__((ext_vector_type(8)));

#define HDIM   128
#define NTI    8      // 128/16 output column tiles
#define KS1    9      // ceil(257/32) k-steps for edge GEMM1 (K padded to 288)
#define KS2    4      // 128/32 k-steps for edge GEMM2
#define EWAVES 4      // waves per edge-kernel block

// LDS budget (bytes):
//   W1 frags : KS1*NTI*32 v16h = 2304*32 = 73728
//   W2 frags : KS2*NTI*32 v16h = 1024*32 = 32768
//   per-wave : (KS1+KS2)*32 v16h = 416*32 = 13312  (A-frag + m1-frag)
//   per-wave misc: 96 dwords = 384
enum { EDGE_SMEM_BYTES =
        (KS1*NTI*32 + KS2*NTI*32 + EWAVES*(KS1+KS2)*32) * 32  // v16h = 32B
      + EWAVES * 96 * 4 };                                     // 161280 B

__device__ __forceinline__ float silu_f(float x) {
  return x * (1.0f / (1.0f + __expf(-x)));
}

// Wave-local LDS producer->consumer fence (CDNA5 split counters).
__device__ __forceinline__ void wave_lds_fence() {
  asm volatile("s_wait_dscnt 0" ::: "memory");
  __builtin_amdgcn_wave_barrier();
}

// ---------------------------------------------------------------------------
// Fused edge kernel: one wave owns a 16-edge tile.
//   ef = [h[row] | h[col] | dist2]  (16 x 288, zero padded)
//   m  = silu(silu(ef @ W1 + b1) @ W2 + b2)
//   atomicAdd(agg[row], m) ; atomicAdd(pos_next[row], diff * tanh(m.cw + cb))
// ---------------------------------------------------------------------------
__global__ __launch_bounds__(EWAVES * 32)
void egnn_edge_kernel(const float* __restrict__ h,
                      const float* __restrict__ pos_cur,
                      float* __restrict__ pos_next,
                      float* __restrict__ agg,
                      const int* __restrict__ ei,      // [2*E] row then col
                      const float* __restrict__ w1,    // [257][128]
                      const float* __restrict__ b1,    // [128]
                      const float* __restrict__ w2,    // [128][128]
                      const float* __restrict__ b2,    // [128]
                      const float* __restrict__ cw,    // [128]
                      const float* __restrict__ cb,    // [1]
                      int E)
{
  extern __shared__ char smem[];
  v16h* w1f   = reinterpret_cast<v16h*>(smem);           // [KS1*NTI*32]
  v16h* w2f   = w1f + KS1 * NTI * 32;                    // [KS2*NTI*32]
  v16h* wbase = w2f + KS2 * NTI * 32;

  const int tid  = threadIdx.x;
  const int lane = tid & 31;
  const int wave = tid >> 5;

  v16h* af  = wbase + wave * ((KS1 + KS2) * 32);         // ef A-frags
  v16h* m1f = af + KS1 * 32;                             // hidden A-frags

  int*   misc  = reinterpret_cast<int*>(wbase + EWAVES * ((KS1 + KS2) * 32))
               + wave * 96;
  int*   rowi  = misc;                                   // [16]
  int*   coli  = misc + 16;                              // [16]
  float* diffL = reinterpret_cast<float*>(misc + 32);    // [16][3]
  float* d2L   = diffL + 48;                             // [16]

  // ---- Stage W1 (K padded 257->288) into B-fragment layout --------------
  // B 32x16 f16 layout: lane = 16*((kr>>3)&1) + (n%16), half index = (kr&7) + (kr>=16 ? 8 : 0).
  for (int idx = tid; idx < 288 * HDIM; idx += blockDim.x) {
    int k = idx >> 7, n = idx & 127;
    float v = (k < 257) ? w1[k * HDIM + n] : 0.0f;
    int s = k >> 5, kr = k & 31;
    int ln = (((kr >> 3) & 1) << 4) + (n & 15);
    ((_Float16*)&w1f[(s * NTI + (n >> 4)) * 32 + ln])[(kr & 7) + ((kr & 16) ? 8 : 0)] = (_Float16)v;
  }
  for (int idx = tid; idx < HDIM * HDIM; idx += blockDim.x) {
    int k = idx >> 7, n = idx & 127;
    float v = w2[k * HDIM + n];
    int s = k >> 5, kr = k & 31;
    int ln = (((kr >> 3) & 1) << 4) + (n & 15);
    ((_Float16*)&w2f[(s * NTI + (n >> 4)) * 32 + ln])[(kr & 7) + ((kr & 16) ? 8 : 0)] = (_Float16)v;
  }
  __syncthreads();

  const float cb0    = cb[0];
  const int   col_lo = lane & 15;
  const int   mg     = lane >> 4;

  const int numTiles = E >> 4;
  const int wstride  = gridDim.x * EWAVES;
  for (int t = blockIdx.x * EWAVES + wave; t < numTiles; t += wstride) {
    const int base = t << 4;

    // ---- per-edge geometry (lanes 0..15, one edge each) -----------------
    if (lane < 16) {
      int e = base + lane;
      int r = ei[e], c = ei[E + e];
      rowi[lane] = r;  coli[lane] = c;
      float dx = pos_cur[r * 3 + 0] - pos_cur[c * 3 + 0];
      float dy = pos_cur[r * 3 + 1] - pos_cur[c * 3 + 1];
      float dz = pos_cur[r * 3 + 2] - pos_cur[c * 3 + 2];
      diffL[lane * 3 + 0] = dx;
      diffL[lane * 3 + 1] = dy;
      diffL[lane * 3 + 2] = dz;
      d2L[lane] = dx * dx + dy * dy + dz * dz;
    }
    wave_lds_fence();

    // ---- gather ef tile into A-fragment layout --------------------------
    // A 16x32 f16 layout: lane = m + 16*g, g=(kr>>3)&1,
    // half index hh = (kr&7) + (kr>=16 ? 8 : 0).
    for (int j = lane; j < 16 * 288; j += 32) {
      int m = j / 288;
      int k = j - m * 288;
      float v;
      if      (k < 128) v = h[(size_t)rowi[m] * HDIM + k];
      else if (k < 256) v = h[(size_t)coli[m] * HDIM + (k - 128)];
      else if (k == 256) v = d2L[m];
      else              v = 0.0f;
      int s = k >> 5, kr = k & 31;
      int g  = (kr >> 3) & 1;
      int hh = (kr & 7) + ((kr & 16) ? 8 : 0);
      ((_Float16*)&af[s * 32 + (g * 16 + m)])[hh] = (_Float16)v;
    }
    wave_lds_fence();

    // ---- GEMM1: [16x288] x [288x128], +b1, SiLU -> m1 A-frags -----------
    for (int nt = 0; nt < NTI; ++nt) {
      v8f acc = {};
#pragma unroll
      for (int s = 0; s < KS1; ++s) {
        acc = __builtin_amdgcn_wmma_f32_16x16x32_f16(
            false, af[s * 32 + lane],
            false, w1f[(s * NTI + nt) * 32 + lane],
            (short)0, acc, false, false);
      }
      int col   = nt * 16 + col_lo;
      float bia = b1[col];
      int s2 = col >> 5, kr2 = col & 31;
      int g2  = (kr2 >> 3) & 1;
      int hh2 = (kr2 & 7) + ((kr2 & 16) ? 8 : 0);
      _Float16* dst = (_Float16*)&m1f[s2 * 32 + g2 * 16];
#pragma unroll
      for (int r = 0; r < 8; ++r) {
        int M = r + 8 * mg;                    // edge within tile
        dst[M * 16 + hh2] = (_Float16)silu_f(acc[r] + bia);
      }
    }
    wave_lds_fence();

    // ---- GEMM2: [16x128] x [128x128], +b2, SiLU; agg atomics; coord -----
    float cp[8];
#pragma unroll
    for (int r = 0; r < 8; ++r) cp[r] = 0.0f;

    for (int nt = 0; nt < NTI; ++nt) {
      v8f acc = {};
#pragma unroll
      for (int s = 0; s < KS2; ++s) {
        acc = __builtin_amdgcn_wmma_f32_16x16x32_f16(
            false, m1f[s * 32 + lane],
            false, w2f[(s * NTI + nt) * 32 + lane],
            (short)0, acc, false, false);
      }
      int col   = nt * 16 + col_lo;
      float bia = b2[col];
      float cwc = cw[col];
#pragma unroll
      for (int r = 0; r < 8; ++r) {
        float v = silu_f(acc[r] + bia);
        cp[r] += v * cwc;
        int M = r + 8 * mg;
        atomicAdd(&agg[(size_t)rowi[M] * HDIM + col], v);
      }
    }

    // reduce coord partials across the 16 lanes of each half-wave
#pragma unroll
    for (int r = 0; r < 8; ++r) {
      cp[r] += __shfl_xor(cp[r], 1, 32);
      cp[r] += __shfl_xor(cp[r], 2, 32);
      cp[r] += __shfl_xor(cp[r], 4, 32);
      cp[r] += __shfl_xor(cp[r], 8, 32);
    }
    if (col_lo == 0) {
#pragma unroll
      for (int r = 0; r < 8; ++r) {
        int   M  = r + 8 * mg;
        float tt = tanhf(cp[r] + cb0);
        int   rn = rowi[M];
        atomicAdd(&pos_next[rn * 3 + 0], diffL[M * 3 + 0] * tt);
        atomicAdd(&pos_next[rn * 3 + 1], diffL[M * 3 + 1] * tt);
        atomicAdd(&pos_next[rn * 3 + 2], diffL[M * 3 + 2] * tt);
      }
    }
    wave_lds_fence();   // protect rowi/diff/af before next tile overwrites
  }
}

// ---------------------------------------------------------------------------
// Small helper kernels (VALU; <3% of total FLOPs)
// ---------------------------------------------------------------------------
__global__ void egnn_embed_kernel(const float* __restrict__ x,
                                  const float* __restrict__ W,
                                  const float* __restrict__ b,
                                  float* __restrict__ h, int Nn) {
  int idx = blockIdx.x * blockDim.x + threadIdx.x;
  if (idx >= Nn * HDIM) return;
  int n = idx >> 7, c = idx & 127;
  float acc = b[c];
#pragma unroll
  for (int k = 0; k < 11; ++k) acc += x[(size_t)n * 11 + k] * W[k * HDIM + c];
  h[idx] = acc;
}

__global__ void egnn_node_kernel(const float* __restrict__ h,
                                 const float* __restrict__ agg,
                                 const float* __restrict__ nw,   // [256][128]
                                 const float* __restrict__ nb,   // [128]
                                 float* __restrict__ hout, int Nn) {
  int idx = blockIdx.x * blockDim.x + threadIdx.x;
  if (idx >= Nn * HDIM) return;
  int n = idx >> 7, c = idx & 127;
  const float* hr = h   + (size_t)n * HDIM;
  const float* ar = agg + (size_t)n * HDIM;
  float acc = nb[c];
  for (int k = 0; k < HDIM; ++k) acc += hr[k] * nw[k * HDIM + c];
  for (int k = 0; k < HDIM; ++k) acc += ar[k] * nw[(HDIM + k) * HDIM + c];
  hout[idx] = silu_f(acc);
}

__global__ void zero_f32_kernel(float* __restrict__ p, size_t n) {
  size_t i = (size_t)blockIdx.x * blockDim.x + threadIdx.x;
  if (i < n) p[i] = 0.0f;
}

__global__ void copy_f32_kernel(const float* __restrict__ s,
                                float* __restrict__ d, size_t n) {
  size_t i = (size_t)blockIdx.x * blockDim.x + threadIdx.x;
  if (i < n) d[i] = s[i];
}

__global__ void egnn_pool_kernel(const float* __restrict__ h,
                                 const int* __restrict__ batch,
                                 float* __restrict__ sums,
                                 float* __restrict__ counts, int Nn) {
  int idx = blockIdx.x * blockDim.x + threadIdx.x;
  if (idx >= Nn * HDIM) return;
  int n = idx >> 7, c = idx & 127;
  int g = batch[n];
  atomicAdd(&sums[(size_t)g * HDIM + c], h[idx]);
  if (c == 0) atomicAdd(&counts[g], 1.0f);
}

__global__ void egnn_head1_kernel(const float* __restrict__ sums,
                                  const float* __restrict__ counts,
                                  const float* __restrict__ w1,  // [128][128]
                                  const float* __restrict__ b1,  // [128]
                                  float* __restrict__ hid, int G) {
  int idx = blockIdx.x * blockDim.x + threadIdx.x;
  if (idx >= G * HDIM) return;
  int g = idx >> 7, c = idx & 127;
  float inv = 1.0f / fmaxf(counts[g], 1.0f);
  const float* sr = sums + (size_t)g * HDIM;
  float acc = b1[c];
  for (int k = 0; k < HDIM; ++k) acc += (sr[k] * inv) * w1[k * HDIM + c];
  hid[idx] = silu_f(acc);
}

__global__ void egnn_head2_kernel(const float* __restrict__ hid,
                                  const float* __restrict__ w2,  // [128]
                                  const float* __restrict__ b2,  // [1]
                                  float* __restrict__ out, int G) {
  int g = blockIdx.x * blockDim.x + threadIdx.x;
  if (g >= G) return;
  const float* hr = hid + (size_t)g * HDIM;
  float acc = b2[0];
  for (int k = 0; k < HDIM; ++k) acc += hr[k] * w2[k];
  out[g] = acc;
}

// ---------------------------------------------------------------------------
extern "C" void kernel_launch(void* const* d_in, const int* in_sizes, int n_in,
                              void* d_out, int out_size, void* d_ws, size_t ws_size,
                              hipStream_t stream) {
  (void)n_in; (void)ws_size;
  const float* x    = (const float*)d_in[0];
  const float* pos  = (const float*)d_in[1];
  const int*   ei   = (const int*)d_in[2];
  const int*   bat  = (const int*)d_in[3];
  const float* embW = (const float*)d_in[4];
  const float* embB = (const float*)d_in[5];
  const float* ew1  = (const float*)d_in[6];
  const float* eb1  = (const float*)d_in[7];
  const float* ew2  = (const float*)d_in[8];
  const float* eb2  = (const float*)d_in[9];
  const float* nw   = (const float*)d_in[10];
  const float* nb   = (const float*)d_in[11];
  const float* cw   = (const float*)d_in[12];
  const float* cb   = (const float*)d_in[13];
  const float* hw1  = (const float*)d_in[14];
  const float* hb1  = (const float*)d_in[15];
  const float* hw2  = (const float*)d_in[16];
  const float* hb2  = (const float*)d_in[17];

  const int Nn = in_sizes[0] / 11;
  const int E  = in_sizes[2] / 2;
  const int G  = out_size;
  const int L  = 4;

  // workspace carve-up (floats)
  const size_t NH = (size_t)Nn * HDIM;
  float* ws     = (float*)d_ws;
  float* hA     = ws;
  float* hB     = hA + NH;
  float* agg    = hB + NH;
  float* posA   = agg + NH;
  float* posB   = posA + (size_t)Nn * 3;
  float* sums   = posB + (size_t)Nn * 3;
  float* counts = sums + (size_t)G * HDIM;
  float* hid    = counts + (size_t)G;

  const int T = 256;
  const int gNH   = (int)((NH + T - 1) / T);
  const int gPos  = (int)(((size_t)Nn * 3 + T - 1) / T);
  const int gGH   = (G * HDIM + T - 1) / T;

  // init: pos copy + embedding
  copy_f32_kernel<<<gPos, T, 0, stream>>>(pos, posA, (size_t)Nn * 3);
  egnn_embed_kernel<<<gNH, T, 0, stream>>>(x, embW, embB, hA, Nn);

  const int edgeBlocks = 640;
  for (int l = 0; l < L; ++l) {
    zero_f32_kernel<<<gNH, T, 0, stream>>>(agg, NH);
    copy_f32_kernel<<<gPos, T, 0, stream>>>(posA, posB, (size_t)Nn * 3);

    egnn_edge_kernel<<<edgeBlocks, EWAVES * 32, EDGE_SMEM_BYTES, stream>>>(
        hA, posA, posB, agg, ei,
        ew1 + (size_t)l * 257 * HDIM, eb1 + (size_t)l * HDIM,
        ew2 + (size_t)l * HDIM * HDIM, eb2 + (size_t)l * HDIM,
        cw + (size_t)l * HDIM, cb + l, E);

    egnn_node_kernel<<<gNH, T, 0, stream>>>(
        hA, agg, nw + (size_t)l * 2 * HDIM * HDIM, nb + (size_t)l * HDIM,
        hB, Nn);

    { float* t = hA; hA = hB; hB = t; }
    { float* t = posA; posA = posB; posB = t; }
  }

  // global mean pool + head
  zero_f32_kernel<<<gGH, T, 0, stream>>>(sums, (size_t)G * HDIM);
  zero_f32_kernel<<<(G + T - 1) / T, T, 0, stream>>>(counts, (size_t)G);
  egnn_pool_kernel<<<gNH, T, 0, stream>>>(hA, bat, sums, counts, Nn);
  egnn_head1_kernel<<<gGH, T, 0, stream>>>(sums, counts, hw1, hb1, hid, G);
  egnn_head2_kernel<<<(G + T - 1) / T, T, 0, stream>>>(hid, hw2, hb2,
                                                       (float*)d_out, G);
}
